// GATSubstAttention_57011395887502
// MI455X (gfx1250) — matrix operand order repaired and checked by reading, not compiled, not run on hardware
//
#include <hip/hip_runtime.h>
#include <hip/hip_bf16.h>

typedef __attribute__((ext_vector_type(16))) _Float16 v16h;
typedef __attribute__((ext_vector_type(8)))  _Float16 v8h;
typedef __attribute__((ext_vector_type(2)))  _Float16 v2h;
typedef __attribute__((ext_vector_type(8)))  float    v8f;

#define N_NODES   50000
#define E_RAW     600000
#define E_TOT     (E_RAW + N_NODES)
#define FEAT      64
#define HID       64
#define HEADS     4
#define NUM_SUBST 32
#define NUM_GRAPHS 128
#define KP1       96            // FEAT+1 padded to multiple of 32
#define NC1       (HEADS*HID)   // 256
#define KP2       256
#define NC2       HID           // 64
#define MT_NODES  (N_NODES/16)  // 3125 exact row-tiles

// A-fragment K-offset within a 32-wide K tile (CDNA5 ISA 7.12.2, 16-bit A):
// lane = kh*16+m ; element e=2v+j ; koff = (v<4 ? kh*8+2v+j : 16+kh*8+2(v-4)+j)
__device__ __forceinline__ int a_koff(int kh, int e) {
    int v = e >> 1, j = e & 1;
    return (v < 4) ? (kh * 8 + 2 * v + j) : (16 + kh * 8 + 2 * (v - 4) + j);
}
// B-fragment: lane = kh*16+nn ; element e ; K = (e<8 ? kh*8+e : 16+kh*8+(e-8)) ; col = nn
__device__ __forceinline__ int b_koff(int kh, int e) {
    return (e < 8) ? (kh * 8 + e) : (16 + kh * 8 + (e - 8));
}

// ---------------------------------------------------------------------------
// Stage 1: per-substructure sums (LDS-staged atomics)
// ---------------------------------------------------------------------------
__global__ void k_subst_acc(const float* __restrict__ x, float* cnt, float* ssum, int n) {
    __shared__ float s_sum[NUM_SUBST * FEAT];
    __shared__ float s_cnt[NUM_SUBST];
    for (int i = threadIdx.x; i < NUM_SUBST * FEAT; i += blockDim.x) s_sum[i] = 0.f;
    if (threadIdx.x < NUM_SUBST) s_cnt[threadIdx.x] = 0.f;
    __syncthreads();
    for (int node = blockIdx.x * blockDim.x + threadIdx.x; node < n;
         node += gridDim.x * blockDim.x) {
        int sid = (int)x[node * FEAT + 5];
        sid = min(max(sid, 0), NUM_SUBST - 1);
        atomicAdd(&s_cnt[sid], 1.0f);
        for (int d = 0; d < FEAT; ++d)
            atomicAdd(&s_sum[sid * FEAT + d], x[node * FEAT + d]);
    }
    __syncthreads();
    for (int i = threadIdx.x; i < NUM_SUBST * FEAT; i += blockDim.x)
        atomicAdd(&ssum[i], s_sum[i]);
    if (threadIdx.x < NUM_SUBST) atomicAdd(&cnt[threadIdx.x], s_cnt[threadIdx.x]);
}

// ---------------------------------------------------------------------------
// Stage 2: substructure MLP + softmax  (one block, 32 threads)
// ---------------------------------------------------------------------------
__global__ void k_subst_mlp(const float* cnt, const float* ssum,
                            const float* w_sa1, const float* b_sa1,
                            const float* w_sa2, const float* b_sa2, float* wout) {
    int s = threadIdx.x;
    float smean[FEAT];
    float c = fmaxf(cnt[s], 1.0f);
    for (int d = 0; d < FEAT; ++d) smean[d] = ssum[s * FEAT + d] / c;
    float logit = b_sa2[0];
    for (int j = 0; j < 32; ++j) {
        float hj = b_sa1[j];
        for (int d = 0; d < FEAT; ++d) hj += smean[d] * w_sa1[d * 32 + j];
        hj = hj > 0.f ? hj : 0.2f * hj;
        logit += hj * w_sa2[j];
    }
    __shared__ float sl[NUM_SUBST];
    sl[s] = logit;
    __syncthreads();
    float mx = -3.4e38f;
    for (int i = 0; i < NUM_SUBST; ++i) mx = fmaxf(mx, sl[i]);
    float ex = __expf(logit - mx);
    __shared__ float se[NUM_SUBST];
    se[s] = ex;
    __syncthreads();
    float sum = 0.f;
    for (int i = 0; i < NUM_SUBST; ++i) sum += se[i];
    wout[s] = ex / sum;
}

// ---------------------------------------------------------------------------
// Pack A1 fragment-major from x + substructure weights.
// Layout: [mt][kt][lane][16 elems] f16, one thread builds one 32B fragment.
// ---------------------------------------------------------------------------
__global__ void k_pack_Afrag1(const float* __restrict__ x, const float* __restrict__ w,
                              _Float16* __restrict__ Ap) {
    const int KT = KP1 / 32;  // 3
    int i = blockIdx.x * blockDim.x + threadIdx.x;
    if (i >= MT_NODES * KT * 32) return;
    int lane = i & 31;
    int kt = (i >> 5) % KT;
    int mt = (i >> 5) / KT;
    int m = lane & 15, kh = lane >> 4;
    int node = mt * 16 + m;                 // always < N_NODES (exact tiling)
    v16h frag;
    #pragma unroll
    for (int e = 0; e < 16; ++e) {
        int K = kt * 32 + a_koff(kh, e);
        float v;
        if (K < FEAT) v = x[node * FEAT + K];
        else if (K == FEAT) {
            int sid = (int)x[node * FEAT + 5];
            sid = min(max(sid, 0), NUM_SUBST - 1);
            v = w[sid];
        } else v = 0.f;
        frag[e] = (_Float16)v;
    }
    *(v16h*)(Ap + (size_t)i * 16) = frag;
}

// ---------------------------------------------------------------------------
// Pack B fragment-major from row-major f32 weights [Ksrc, NC] (K zero-padded).
// Layout: [kt][nt][lane][16 elems] f16.
// ---------------------------------------------------------------------------
__global__ void k_pack_Bfrag(const float* __restrict__ W, _Float16* __restrict__ Bp,
                             int Ksrc, int Kp, int NC) {
    int KT = Kp / 32, NT = NC / 16;
    int i = blockIdx.x * blockDim.x + threadIdx.x;
    if (i >= KT * NT * 32) return;
    int lane = i & 31;
    int nt = (i >> 5) % NT;
    int kt = (i >> 5) / NT;
    int nn = lane & 15, kh = lane >> 4;
    int col = nt * 16 + nn;
    v16h frag;
    #pragma unroll
    for (int e = 0; e < 16; ++e) {
        int K = kt * 32 + b_koff(kh, e);
        frag[e] = (K < Ksrc) ? (_Float16)W[(size_t)K * NC + col] : (_Float16)0.f;
    }
    *(v16h*)(Bp + (size_t)i * 16) = frag;
}

// ---------------------------------------------------------------------------
// WMMA GEMM on pre-packed fragments.  256 threads = 8 waves; each wave owns a
// 16x64 strip (4 N-tiles, A-fragment reused across them).  All layout math is
// constexpr -> immediate-offset b128 loads feeding v_wmma.
// ---------------------------------------------------------------------------
template <int KP, int NC, int CG>
__global__ __launch_bounds__(256)
void k_gemm_wmma(const _Float16* __restrict__ Ap, const _Float16* __restrict__ Bp,
                 float* __restrict__ C, int M) {
    constexpr int KT = KP / 32;
    constexpr int NT = NC / 16;
    constexpr int ROWT = 8 / CG;           // row tiles per block
    const int lane = threadIdx.x & 31;
    const int wave = threadIdx.x >> 5;
    const int cg = wave % CG;
    const int rt = wave / CG;
    const int rowBase = blockIdx.x * (16 * ROWT) + rt * 16;
    const int colBase = cg * 64;
    const int mtMax = (M >> 4) - 1;
    const int mt = min(rowBase >> 4, mtMax);   // clamp keeps EXEC full in loop

    const v16h* Af = (const v16h*)Ap + (size_t)mt * KT * 32 + lane;
    const v16h* Bf = (const v16h*)Bp + lane;

    v8f c0 = {}, c1 = {}, c2 = {}, c3 = {};
    #pragma unroll
    for (int kt = 0; kt < KT; ++kt) {
        v16h a  = Af[kt * 32];
        v16h b0 = Bf[(kt * NT + cg * 4 + 0) * 32];
        v16h b1 = Bf[(kt * NT + cg * 4 + 1) * 32];
        v16h b2 = Bf[(kt * NT + cg * 4 + 2) * 32];
        v16h b3 = Bf[(kt * NT + cg * 4 + 3) * 32];
        c0 = __builtin_amdgcn_wmma_f32_16x16x32_f16(false, a, false, b0, (short)0, c0, false, false);
        c1 = __builtin_amdgcn_wmma_f32_16x16x32_f16(false, a, false, b1, (short)0, c1, false, false);
        c2 = __builtin_amdgcn_wmma_f32_16x16x32_f16(false, a, false, b2, (short)0, c2, false, false);
        c3 = __builtin_amdgcn_wmma_f32_16x16x32_f16(false, a, false, b3, (short)0, c3, false, false);
    }
    const int nn = lane & 15, kh = lane >> 4;
    #pragma unroll
    for (int r = 0; r < 8; ++r) {
        int mrow = rowBase + r + 8 * kh;     // C: VGPR r holds M = r + 8*kh
        if (mrow < M) {
            float* cp = C + (size_t)mrow * NC + colBase + nn;
            cp[0]  = c0[r];
            cp[16] = c1[r];
            cp[32] = c2[r];
            cp[48] = c3[r];
        }
    }
}

// ---------------------------------------------------------------------------
// Attention coefficients: as/ad [n,H] = <h[n,h,:], att>
// ---------------------------------------------------------------------------
__global__ void k_att(const float* __restrict__ h, const float* __restrict__ att_s,
                      const float* __restrict__ att_d, float* as_, float* ad_,
                      int n, int H, int D) {
    int i = blockIdx.x * blockDim.x + threadIdx.x;
    if (i >= n * H) return;
    int node = i / H, hh = i % H;
    const float* hp = h + (size_t)node * H * D + hh * D;
    float s = 0.f, d = 0.f;
    for (int k = 0; k < D; ++k) { s += hp[k] * att_s[hh * D + k]; d += hp[k] * att_d[hh * D + k]; }
    as_[i] = s; ad_[i] = d;
}

__device__ __forceinline__ unsigned f2mono(float f) {
    unsigned u = __float_as_uint(f);
    return (u & 0x80000000u) ? ~u : (u | 0x80000000u);
}
__device__ __forceinline__ float mono2f(unsigned u) {
    return __uint_as_float((u & 0x80000000u) ? (u & 0x7FFFFFFFu) : ~u);
}

// ---------------------------------------------------------------------------
// Edge pass 1: e = leaky(as[src]+ad[dst]); segment max per dst (monotone-uint)
// ---------------------------------------------------------------------------
__global__ void k_edge1(const int* __restrict__ ei, const float* __restrict__ as_,
                        const float* __restrict__ ad_, float* __restrict__ ebuf,
                        unsigned* __restrict__ mEnc, int H) {
    int i = blockIdx.x * blockDim.x + threadIdx.x;
    if (i >= E_TOT * H) return;
    int e = i / H, hh = i % H;
    int src = (e < E_RAW) ? ei[e] : (e - E_RAW);
    int dst = (e < E_RAW) ? ei[E_RAW + e] : (e - E_RAW);
    float ev = as_[src * H + hh] + ad_[dst * H + hh];
    ev = ev > 0.f ? ev : 0.2f * ev;
    ebuf[i] = ev;
    atomicMax(&mEnc[dst * H + hh], f2mono(ev));
}

// ---------------------------------------------------------------------------
// Edge pass 2: ex = exp(e-m); den[dst]+=ex; num[dst,:]+=ex*h[src,:]
// ---------------------------------------------------------------------------
__global__ void k_edge2(const int* __restrict__ ei, const float* __restrict__ ebuf,
                        const unsigned* __restrict__ mEnc, const float* __restrict__ h,
                        float* __restrict__ den, float* __restrict__ num, int H, int D) {
    int i = blockIdx.x * blockDim.x + threadIdx.x;
    if (i >= E_TOT * H) return;
    int e = i / H, hh = i % H;
    int src = (e < E_RAW) ? ei[e] : (e - E_RAW);
    int dst = (e < E_RAW) ? ei[E_RAW + e] : (e - E_RAW);
    float m = mono2f(mEnc[dst * H + hh]);
    float ex = __expf(ebuf[i] - m);
    atomicAdd(&den[dst * H + hh], ex);
    const float* hp = h + (size_t)src * H * D + hh * D;
    float* np = num + (size_t)dst * H * D + hh * D;
    for (int k = 0; k < D; ++k) atomicAdd(&np[k], ex * hp[k]);
}

// ---------------------------------------------------------------------------
// Finalize layer 1: ÷denom, +bias, ELU -> write DIRECTLY in packed A-fragment
// layout for GEMM2 (f16, [mt][kt][lane][16]).  One thread = 2 consecutive K.
// ---------------------------------------------------------------------------
__global__ void k_fin1_pack(const float* __restrict__ num, const float* __restrict__ den,
                            const float* __restrict__ bias, _Float16* __restrict__ Ap,
                            int n) {
    constexpr int KT = KP2 / 32;   // 8
    int i = blockIdx.x * blockDim.x + threadIdx.x;
    if (i >= n * (NC1 / 2)) return;
    int node = i / (NC1 / 2);
    int p = i % (NC1 / 2);
    int c0 = 2 * p;                     // even channel, c1 = c0+1
    int hh = c0 / HID;
    float dv = den[node * HEADS + hh] + 1e-16f;
    float v0 = num[(size_t)node * NC1 + c0]     / dv + bias[c0];
    float v1 = num[(size_t)node * NC1 + c0 + 1] / dv + bias[c0 + 1];
    v0 = v0 > 0.f ? v0 : (__expf(v0) - 1.0f);
    v1 = v1 > 0.f ? v1 : (__expf(v1) - 1.0f);
    // inverse A-layout mapping: ko -> (kh, e)
    int kt = c0 >> 5;
    int ko = c0 & 31;
    int group = ko >> 3;                // 0..3
    int r = ko & 7;
    int kh = group & 1;
    int e = (group < 2) ? r : (8 + r);  // even
    int mt = node >> 4, m = node & 15;
    int lane = kh * 16 + m;
    size_t idx = (((size_t)mt * KT + kt) * 32 + lane) * 16 + e;
    v2h pr; pr[0] = (_Float16)v0; pr[1] = (_Float16)v1;
    *(v2h*)(Ap + idx) = pr;
}

// ---------------------------------------------------------------------------
// Finalize layer 2 + global_add_pool
// ---------------------------------------------------------------------------
__global__ void k_fin2_pool(const float* __restrict__ num, const float* __restrict__ den,
                            const float* __restrict__ bias, const int* __restrict__ batch,
                            float* __restrict__ g, int n, int D) {
    int i = blockIdx.x * blockDim.x + threadIdx.x;
    if (i >= n * D) return;
    int node = i / D, c = i % D;
    float v = num[i] / (den[node] + 1e-16f) + bias[c];
    atomicAdd(&g[batch[node] * D + c], v);
}

// ---------------------------------------------------------------------------
// Prediction head: out = elu(g@Wp1+bp1)@Wp2+bp2   (128 threads, 1 block)
// ---------------------------------------------------------------------------
__global__ void k_head(const float* __restrict__ g, const float* __restrict__ Wp1,
                       const float* __restrict__ bp1, const float* __restrict__ Wp2,
                       const float* __restrict__ bp2, float* __restrict__ out) {
    int i = threadIdx.x;
    if (i >= NUM_GRAPHS) return;
    float acc = bp2[0];
    for (int j = 0; j < HID / 2; ++j) {
        float hj = bp1[j];
        for (int d = 0; d < HID; ++d) hj += g[i * HID + d] * Wp1[d * (HID / 2) + j];
        hj = hj > 0.f ? hj : (__expf(hj) - 1.0f);
        acc += hj * Wp2[j];
    }
    out[i] = acc;
}

// ---------------------------------------------------------------------------
extern "C" void kernel_launch(void* const* d_in, const int* in_sizes, int n_in,
                              void* d_out, int out_size, void* d_ws, size_t ws_size,
                              hipStream_t stream) {
    const float* x      = (const float*)d_in[0];
    const int*   ei     = (const int*)d_in[1];
    const int*   batch  = (const int*)d_in[2];
    const float* w_sa1  = (const float*)d_in[3];
    const float* b_sa1  = (const float*)d_in[4];
    const float* w_sa2  = (const float*)d_in[5];
    const float* b_sa2  = (const float*)d_in[6];
    const float* W1     = (const float*)d_in[7];
    const float* att_s1 = (const float*)d_in[8];
    const float* att_d1 = (const float*)d_in[9];
    const float* b1     = (const float*)d_in[10];
    const float* W2     = (const float*)d_in[11];
    const float* att_s2 = (const float*)d_in[12];
    const float* att_d2 = (const float*)d_in[13];
    const float* b2     = (const float*)d_in[14];
    const float* Wp1    = (const float*)d_in[15];
    const float* bp1    = (const float*)d_in[16];
    const float* Wp2    = (const float*)d_in[17];
    const float* bp2    = (const float*)d_in[18];
    float* out = (float*)d_out;

    // ---- workspace carve-up (256B aligned) ----
    char* base = (char*)d_ws;
    size_t off = 0;
    auto alloc = [&](size_t bytes) -> char* {
        char* p = base + off;
        off = (off + bytes + 255) & ~(size_t)255;
        return p;
    };
    float*    cnt  = (float*)alloc(NUM_SUBST * 4);
    float*    ssum = (float*)alloc(NUM_SUBST * FEAT * 4);
    float*    wS   = (float*)alloc(NUM_SUBST * 4);
    _Float16* A1p  = (_Float16*)alloc((size_t)MT_NODES * (KP1 / 32) * 32 * 16 * 2);
    _Float16* B1p  = (_Float16*)alloc((size_t)(KP1 / 32) * (NC1 / 16) * 32 * 16 * 2);
    _Float16* B2p  = (_Float16*)alloc((size_t)(KP2 / 32) * (NC2 / 16) * 32 * 16 * 2);
    float*    h1   = (float*)alloc((size_t)N_NODES * NC1 * 4);
    float*    as1  = (float*)alloc((size_t)N_NODES * HEADS * 4);
    float*    ad1  = (float*)alloc((size_t)N_NODES * HEADS * 4);
    unsigned* m1   = (unsigned*)alloc((size_t)N_NODES * HEADS * 4);
    float*    den1 = (float*)alloc((size_t)N_NODES * HEADS * 4);
    float*    e1   = (float*)alloc((size_t)E_TOT * HEADS * 4);
    float*    num1 = (float*)alloc((size_t)N_NODES * NC1 * 4);
    _Float16* A2p  = (_Float16*)alloc((size_t)MT_NODES * (KP2 / 32) * 32 * 16 * 2);
    float*    h2   = (float*)alloc((size_t)N_NODES * NC2 * 4);
    float*    as2  = (float*)alloc((size_t)N_NODES * 4);
    float*    ad2  = (float*)alloc((size_t)N_NODES * 4);
    unsigned* m2   = (unsigned*)alloc((size_t)N_NODES * 4);
    float*    den2 = (float*)alloc((size_t)N_NODES * 4);
    float*    e2   = (float*)alloc((size_t)E_TOT * 4);
    float*    num2 = (float*)alloc((size_t)N_NODES * NC2 * 4);
    float*    g    = (float*)alloc((size_t)NUM_GRAPHS * HID * 4);
    (void)ws_size; (void)in_sizes; (void)n_in; (void)out_size;

    // ---- zero-init accumulators ----
    hipMemsetAsync(cnt,  0, NUM_SUBST * 4, stream);
    hipMemsetAsync(ssum, 0, NUM_SUBST * FEAT * 4, stream);
    hipMemsetAsync(m1,   0, (size_t)N_NODES * HEADS * 4, stream);
    hipMemsetAsync(den1, 0, (size_t)N_NODES * HEADS * 4, stream);
    hipMemsetAsync(num1, 0, (size_t)N_NODES * NC1 * 4, stream);
    hipMemsetAsync(m2,   0, (size_t)N_NODES * 4, stream);
    hipMemsetAsync(den2, 0, (size_t)N_NODES * 4, stream);
    hipMemsetAsync(num2, 0, (size_t)N_NODES * NC2 * 4, stream);
    hipMemsetAsync(g,    0, (size_t)NUM_GRAPHS * HID * 4, stream);

    const int TB = 256;
    // substructure gate
    k_subst_acc<<<256, TB, 0, stream>>>(x, cnt, ssum, N_NODES);
    k_subst_mlp<<<1, NUM_SUBST, 0, stream>>>(cnt, ssum, w_sa1, b_sa1, w_sa2, b_sa2, wS);

    // fragment packing
    {
        int nA1 = MT_NODES * (KP1 / 32) * 32;
        k_pack_Afrag1<<<(nA1 + TB - 1) / TB, TB, 0, stream>>>(x, wS, A1p);
        int nB1 = (KP1 / 32) * (NC1 / 16) * 32;
        k_pack_Bfrag<<<(nB1 + TB - 1) / TB, TB, 0, stream>>>(W1, B1p, FEAT + 1, KP1, NC1);
        int nB2 = (KP2 / 32) * (NC2 / 16) * 32;
        k_pack_Bfrag<<<(nB2 + TB - 1) / TB, TB, 0, stream>>>(W2, B2p, KP2, KP2, NC2);
    }

    // GEMM1: [N,96] @ [96,256] -> h1.  CG=4 -> 32 rows/block.
    k_gemm_wmma<KP1, NC1, 4><<<(N_NODES + 31) / 32, TB, 0, stream>>>(A1p, B1p, h1, N_NODES);

    // GAT layer 1
    k_att<<<((size_t)N_NODES * HEADS + TB - 1) / TB, TB, 0, stream>>>(h1, att_s1, att_d1, as1, ad1, N_NODES, HEADS, HID);
    k_edge1<<<((size_t)E_TOT * HEADS + TB - 1) / TB, TB, 0, stream>>>(ei, as1, ad1, e1, m1, HEADS);
    k_edge2<<<((size_t)E_TOT * HEADS + TB - 1) / TB, TB, 0, stream>>>(ei, e1, m1, h1, den1, num1, HEADS, HID);
    k_fin1_pack<<<((size_t)N_NODES * (NC1 / 2) + TB - 1) / TB, TB, 0, stream>>>(num1, den1, b1, A2p, N_NODES);

    // GEMM2: [N,256] @ [256,64] -> h2.  CG=1 -> 128 rows/block.
    k_gemm_wmma<KP2, NC2, 1><<<(N_NODES + 127) / 128, TB, 0, stream>>>(A2p, B2p, h2, N_NODES);

    // GAT layer 2 (1 head)
    k_att<<<((size_t)N_NODES + TB - 1) / TB, TB, 0, stream>>>(h2, att_s2, att_d2, as2, ad2, N_NODES, 1, HID);
    k_edge1<<<((size_t)E_TOT + TB - 1) / TB, TB, 0, stream>>>(ei, as2, ad2, e2, m2, 1);
    k_edge2<<<((size_t)E_TOT + TB - 1) / TB, TB, 0, stream>>>(ei, e2, m2, h2, den2, num2, 1, HID);
    k_fin2_pool<<<((size_t)N_NODES * NC2 + TB - 1) / TB, TB, 0, stream>>>(num2, den2, b2, batch, g, N_NODES, HID);

    // prediction head
    k_head<<<1, NUM_GRAPHS, 0, stream>>>(g, Wp1, bp1, Wp2, bp2, out);
}
